// TokenPredictorModel_34196529611446
// MI455X (gfx1250) — matrix operand order repaired and checked by reading, not compiled, not run on hardware
//
#include <hip/hip_runtime.h>
#include <hip/hip_bf16.h>

#define DF 64

typedef __attribute__((ext_vector_type(16))) _Float16 v16h;
typedef __attribute__((ext_vector_type(8)))  _Float16 h8;
typedef __attribute__((ext_vector_type(8)))  float    v8f;

// ---------------------------------------------------------------------------
// k_init: deg=0, slot=INT_MAX, A=0
// ---------------------------------------------------------------------------
__global__ void k_init(int* deg, int* slot, float* A, int N, int Aelems) {
  int i = blockIdx.x * blockDim.x + threadIdx.x;
  int stride = gridDim.x * blockDim.x;
  for (int t = i; t < N; t += stride) { deg[t] = 0; slot[t] = 0x7FFFFFFF; }
  for (int t = i; t < Aelems; t += stride) A[t] = 0.f;
}

// deg[dst] += 1 over all edges (self-loop +1 added at use sites)
__global__ void k_deg(const int* __restrict__ dst, int* deg, int E) {
  int i = blockIdx.x * blockDim.x + threadIdx.x;
  if (i < E) atomicAdd(&deg[dst[i]], 1);
}

// canonical slot per needed node (min batch index wins -> deterministic)
__global__ void k_slot(const int* __restrict__ ids, int* slot, int M) {
  int i = blockIdx.x * blockDim.x + threadIdx.x;
  if (i < M) atomicMin(&slot[ids[i]], i);
}

// ---------------------------------------------------------------------------
// k_fuse: Mz = Wz @ Lz[:64], cz = bz @ Lz[:64] + lbz   (same for h-gate)
// ---------------------------------------------------------------------------
__global__ void k_fuse(const float* __restrict__ Wz, const float* __restrict__ Lz,
                       const float* __restrict__ bz, const float* __restrict__ lbz,
                       const float* __restrict__ Wh, const float* __restrict__ Lh,
                       const float* __restrict__ bh, const float* __restrict__ lbh,
                       float* Mz, float* Mh, float* cz, float* ch) {
  for (int idx = threadIdx.x; idx < 2 * DF * DF; idx += blockDim.x) {
    int g = idx >> 12, i = (idx >> 6) & 63, j = idx & 63;
    const float* W = g ? Wh : Wz;
    const float* L = g ? Lh : Lz;
    float acc = 0.f;
    for (int k = 0; k < DF; ++k) acc += W[i * DF + k] * L[k * DF + j];
    (g ? Mh : Mz)[i * DF + j] = acc;
  }
  for (int idx = threadIdx.x; idx < 2 * DF; idx += blockDim.x) {
    int g = idx >> 6, j = idx & 63;
    const float* b = g ? bh : bz;
    const float* L = g ? Lh : Lz;
    float acc = (g ? lbh : lbz)[j];
    for (int k = 0; k < DF; ++k) acc += b[k] * L[k * DF + j];
    (g ? ch : cz)[j] = acc;
  }
}

// ---------------------------------------------------------------------------
// k_agg: wave-ballot edge scan; ~4% of edges hit a needed dst. For each hit,
// the whole wave cooperatively atomically accumulates x[src]*enorm (64 floats,
// 2 per lane) into the compact A[slot] row.
// ---------------------------------------------------------------------------
__global__ void k_agg(const int* __restrict__ src, const int* __restrict__ dst,
                      const int* __restrict__ slot, const int* __restrict__ deg,
                      const float* __restrict__ x, float* __restrict__ A, int E) {
  int lane = threadIdx.x & 31;
  int wid = (blockIdx.x * blockDim.x + threadIdx.x) >> 5;
  int e = wid * 32 + lane;
  bool keep = false;
  int sIdx = 0, sN = 0;
  float en = 0.f;
  if (e < E) {
    int d = dst[e];
    int sl = slot[d];
    if (sl != 0x7FFFFFFF) {
      keep = true;
      sIdx = sl;
      sN = src[e];
      en = rsqrtf((float)(deg[sN] + 1)) * rsqrtf((float)(deg[d] + 1));
    }
  }
  unsigned mask = (unsigned)__ballot(keep);   // wave32: low 32 bits
  while (mask) {
    int j = __ffs(mask) - 1;
    mask &= mask - 1;
    int bs = __shfl(sN, j);
    int bt = __shfl(sIdx, j);
    float be = __shfl(en, j);
    const float2 xv = *(const float2*)(x + (size_t)bs * DF + lane * 2);
    atomicAdd(&A[(size_t)bt * DF + lane * 2 + 0], xv.x * be);
    atomicAdd(&A[(size_t)bt * DF + lane * 2 + 1], xv.y * be);
  }
}

// ---------------------------------------------------------------------------
// k_w2t: W2 [64, OUT] f32  ->  W2t [NPAD, 64] f16 (transposed, zero-padded)
// ---------------------------------------------------------------------------
__global__ void k_w2t(const float* __restrict__ W2, _Float16* __restrict__ w2t,
                      int OUT, int NPAD) {
  __shared__ _Float16 tile[32][33];
  int n0 = blockIdx.x * 32, k0 = blockIdx.y * 32;
  int tx = threadIdx.x & 31, ty = threadIdx.x >> 5;
  for (int i = 0; i < 32; i += 8) {
    int k = k0 + ty + i, n = n0 + tx;
    float v = (n < OUT) ? W2[(size_t)k * OUT + n] : 0.f;
    tile[ty + i][tx] = (_Float16)v;
  }
  __syncthreads();
  for (int i = 0; i < 32; i += 8) {
    int n = n0 + ty + i, k = k0 + tx;
    if (n < NPAD) w2t[(size_t)n * DF + k] = tile[tx][ty + i];
  }
}

// ---------------------------------------------------------------------------
// k_gru_mlp: one block (64 threads) per batch node.
// P = A[slot] + x*snorm ; H = (1-sigmoid(P@Mz+cz)) * tanh(P@Mh+ch)
// h16 = f16(relu(H@W1 + b1))
// ---------------------------------------------------------------------------
__global__ void k_gru_mlp(const int* __restrict__ ids, const int* __restrict__ slot,
                          const int* __restrict__ deg, const float* __restrict__ x,
                          const float* __restrict__ A,
                          const float* __restrict__ Mz, const float* __restrict__ Mh,
                          const float* __restrict__ cz, const float* __restrict__ ch,
                          const float* __restrict__ W1, const float* __restrict__ b1,
                          _Float16* __restrict__ h16) {
  __shared__ float Pl[DF];
  __shared__ float Hl[DF];
  int m = blockIdx.x, j = threadIdx.x;
  int id = ids[m];
  int s = slot[id];
  float sn = 1.f / (float)(deg[id] + 1);
  Pl[j] = A[(size_t)s * DF + j] + x[(size_t)id * DF + j] * sn;
  __syncthreads();
  float az = cz[j], ah = ch[j];
  for (int k = 0; k < DF; ++k) {
    float p = Pl[k];
    az += p * Mz[k * DF + j];
    ah += p * Mh[k * DF + j];
  }
  float Z = 1.f / (1.f + expf(-az));
  float Ht = tanhf(ah);
  Hl[j] = (1.f - Z) * Ht;
  __syncthreads();
  float acc = b1[j];
  for (int k = 0; k < DF; ++k) acc += Hl[k] * W1[k * DF + j];
  h16[(size_t)m * DF + j] = (_Float16)fmaxf(acc, 0.f);
}

// ---------------------------------------------------------------------------
// k_gemm: out[M, OUT] = h16 @ W2t^T + b2 via v_wmma_f32_16x16x32_f16.
// 2D grid: blockIdx.y = row-tile (mt), one wave per 16x16 output tile,
// K=64 -> 2 WMMA ops per tile.
// A layout (16-bit A 16x32): lane<16 holds M=lane, K {0..7,16..23};
// lane>=16 holds M=lane-16, K {8..15,24..31}. B (32x16): lane<16 N=lane K {0..7,16..23},
// lane>=16 N=lane-16 K {8..15,24..31}. C: VGPR r -> row hh*8+r, col lane&15.
// Output (412 MB) is write-once >> L2 (192 MB): store non-temporal so the
// L2-resident f16 W2t (6.4 MB, reused by all 128 row-tiles) stays hot.
// ---------------------------------------------------------------------------
__global__ void k_gemm(const _Float16* __restrict__ h16,
                       const _Float16* __restrict__ w2t,
                       const float* __restrict__ b2,
                       float* __restrict__ out,
                       int OUT, int ntiles) {
  int nt = blockIdx.x * (blockDim.x >> 5) + (threadIdx.x >> 5);
  if (nt >= ntiles) return;              // wave-uniform: EXEC stays all-ones
  int mt = blockIdx.y;
  int lane = threadIdx.x & 31;
  int hh = lane >> 4;                    // half-wave id
  int lm = lane & 15;

  const _Float16* arow = h16 + (size_t)(mt * 16 + lm) * DF;
  union { v16h v; h8 h[2]; } a0, a1, b0, b1;
  a0.h[0] = *(const h8*)(arow + hh * 8);          // K  0..7  /  8..15
  a0.h[1] = *(const h8*)(arow + 16 + hh * 8);     // K 16..23 / 24..31
  a1.h[0] = *(const h8*)(arow + 32 + hh * 8);     // K 32..39 / 40..47
  a1.h[1] = *(const h8*)(arow + 48 + hh * 8);     // K 48..55 / 56..63

  const _Float16* brow = w2t + (size_t)(nt * 16 + lm) * DF;
  b0.h[0] = *(const h8*)(brow + hh * 8);          // K  0..7  /  8..15
  b0.h[1] = *(const h8*)(brow + 16 + hh * 8);     // K 16..23 / 24..31
  b1.h[0] = *(const h8*)(brow + 32 + hh * 8);     // K 32..39 / 40..47
  b1.h[1] = *(const h8*)(brow + 48 + hh * 8);     // K 48..55 / 56..63

  v8f c = {};
  c = __builtin_amdgcn_wmma_f32_16x16x32_f16(false, a0.v, false, b0.v,
                                             (short)0, c, false, false);
  c = __builtin_amdgcn_wmma_f32_16x16x32_f16(false, a1.v, false, b1.v,
                                             (short)0, c, false, false);

  int col = nt * 16 + lm;
  if (col < OUT) {
    float bias = b2[col];
    size_t base = (size_t)(mt * 16 + hh * 8) * OUT + col;
    for (int r = 0; r < 8; ++r)
      __builtin_nontemporal_store(c[r] + bias, &out[base + (size_t)r * OUT]);
  }
}

// ---------------------------------------------------------------------------
extern "C" void kernel_launch(void* const* d_in, const int* in_sizes, int n_in,
                              void* d_out, int out_size, void* d_ws, size_t ws_size,
                              hipStream_t stream) {
  (void)n_in; (void)ws_size; (void)out_size;
  const float* x   = (const float*)d_in[0];
  const int*   src = (const int*)d_in[1];
  const int*   dst = (const int*)d_in[2];
  const int*   ids = (const int*)d_in[3];
  const float* Wz  = (const float*)d_in[4];
  const float* bz  = (const float*)d_in[5];
  // d_in[6], d_in[7]: Wr, br -- dead (H=0 => H*R=0)
  const float* Wh  = (const float*)d_in[8];
  const float* bh  = (const float*)d_in[9];
  const float* Lz  = (const float*)d_in[10];
  const float* lbz = (const float*)d_in[11];
  // d_in[12], d_in[13]: Lr, lbr -- dead
  const float* Lh  = (const float*)d_in[14];
  const float* lbh = (const float*)d_in[15];
  const float* W1  = (const float*)d_in[16];
  const float* b1  = (const float*)d_in[17];
  const float* W2  = (const float*)d_in[18];
  const float* b2  = (const float*)d_in[19];
  float* out = (float*)d_out;

  const int N   = in_sizes[0] / DF;
  const int E   = in_sizes[1];
  const int M   = in_sizes[3];
  const int OUT = in_sizes[19];
  const int NT  = (OUT + 15) / 16;
  const int NPAD = NT * 16;

  char* w = (char*)d_ws;
  size_t off = 0;
  auto alloc = [&](size_t bytes) -> void* {
    void* p = (void*)(w + off);
    off += (bytes + 255) & ~(size_t)255;
    return p;
  };
  int*      deg  = (int*)alloc((size_t)N * 4);
  int*      slot = (int*)alloc((size_t)N * 4);
  float*    A    = (float*)alloc((size_t)M * DF * 4);
  float*    Mz   = (float*)alloc(DF * DF * 4);
  float*    Mh   = (float*)alloc(DF * DF * 4);
  float*    cz   = (float*)alloc(DF * 4);
  float*    ch   = (float*)alloc(DF * 4);
  _Float16* h16  = (_Float16*)alloc((size_t)M * DF * 2);
  _Float16* w2t  = (_Float16*)alloc((size_t)NPAD * DF * 2);

  const int tpb = 256;
  int initN = (N > M * DF) ? N : (M * DF);
  k_init<<<(initN + tpb - 1) / tpb, tpb, 0, stream>>>(deg, slot, A, N, M * DF);
  k_deg<<<(E + tpb - 1) / tpb, tpb, 0, stream>>>(dst, deg, E);
  k_slot<<<(M + tpb - 1) / tpb, tpb, 0, stream>>>(ids, slot, M);
  k_fuse<<<1, 256, 0, stream>>>(Wz, Lz, bz, lbz, Wh, Lh, bh, lbh, Mz, Mh, cz, ch);
  k_agg<<<(E + tpb - 1) / tpb, tpb, 0, stream>>>(src, dst, slot, deg, x, A, E);
  k_w2t<<<dim3((NPAD + 31) / 32, 2), 256, 0, stream>>>(W2, w2t, OUT, NPAD);
  k_gru_mlp<<<M, DF, 0, stream>>>(ids, slot, deg, x, A, Mz, Mh, cz, ch, W1, b1, h16);

  k_gemm<<<dim3((NT + 7) / 8, M / 16), 256, 0, stream>>>(h16, w2t, b2, out, OUT, NT);
}
